// MultilayersGAT_9174050144907
// MI455X (gfx1250) — matrix operand order repaired and checked by reading, not compiled, not run on hardware
//
#include <hip/hip_runtime.h>
#include <cstdint>
#include <cstddef>

// ---------------- problem constants ----------------
#define GN    2048      // nodes
#define FIN   256       // input features (all layers: 4 heads * 64 = 256)
#define HID   64        // per-head hidden
#define NH    4         // heads
#define NL    3         // layers
#define EF    32        // edge features
#define GALPHA 0.2f
#define PS    ((size_t)GN * (size_t)GN)   // e_score plane stride (elements)

// ---------------- vector types ----------------
typedef __attribute__((ext_vector_type(16))) __bf16          v16bf;
typedef __attribute__((ext_vector_type(8)))  __bf16          v8bf;
typedef __attribute__((ext_vector_type(16))) unsigned short  v16u;
typedef __attribute__((ext_vector_type(8)))  unsigned short  v8u;
typedef __attribute__((ext_vector_type(8)))  float           v8f;

// ---------------- bf16 helpers ----------------
// Use the native f32->bf16 convert (lowers to v_cvt_*bf16* on gfx1250)
__device__ __forceinline__ unsigned short f2bf(float f) {
    __bf16 b = (__bf16)f;
    return __builtin_bit_cast(unsigned short, b);
}
__device__ __forceinline__ float bf2f(unsigned short h) {
    union { unsigned u; float f; } v; v.u = ((unsigned)h) << 16;
    return v.f;
}

__device__ __forceinline__ v8f wmma_bf16(v16bf a, v16bf b, v8f c) {
    // D = A(16x32 bf16) * B(32x16 bf16) + C(16x16 f32)
    return __builtin_amdgcn_wmma_f32_16x16x32_bf16(
        /*neg_a=*/false, a, /*neg_b=*/false, b,
        /*c_mod=*/(short)0, c, /*reuse_a=*/false, /*reuse_b=*/false);
}

// Pack a 16-element A/B fragment slice from 4 float4s (hardware f32 -> bf16).
__device__ __forceinline__ v16bf pack16(const float4 p0, const float4 p1,
                                        const float4 q0, const float4 q1) {
    v16bf u;
    u[0] = (__bf16)p0.x; u[1] = (__bf16)p0.y; u[2]  = (__bf16)p0.z; u[3]  = (__bf16)p0.w;
    u[4] = (__bf16)p1.x; u[5] = (__bf16)p1.y; u[6]  = (__bf16)p1.z; u[7]  = (__bf16)p1.w;
    u[8] = (__bf16)q0.x; u[9] = (__bf16)q0.y; u[10] = (__bf16)q0.z; u[11] = (__bf16)q0.w;
    u[12] = (__bf16)q1.x;u[13] = (__bf16)q1.y;u[14] = (__bf16)q1.z; u[15] = (__bf16)q1.w;
    return u;
}

// =====================================================================
// coef[c][f] = sum_i W_e[c][f][i] * (sum_o W[c][i][o] * a3[c][o]),  c = l*NH+h
// grid: 12 blocks of 256 threads
// =====================================================================
__global__ void k_coef(const float* __restrict__ W, const float* __restrict__ We,
                       const float* __restrict__ a, float* __restrict__ coef) {
    const int c = blockIdx.x;                 // 0..11
    const int tid = threadIdx.x;              // 0..255 == FIN
    __shared__ float t[FIN];
    const float* a3 = a + (size_t)c * 192 + 128;
    const float* wr = W + ((size_t)c * FIN + tid) * HID;
    float s = 0.f;
    #pragma unroll 8
    for (int o = 0; o < HID; ++o) s += wr[o] * a3[o];
    t[tid] = s;
    __syncthreads();
    if (tid < EF) {
        const float* wer = We + ((size_t)c * EF + tid) * FIN;
        float cf = 0.f;
        #pragma unroll 8
        for (int i = 0; i < FIN; ++i) cf += wer[i] * t[i];
        coef[c * EF + tid] = cf;
    }
}

// =====================================================================
// wT[(c*HID+o)*FIN+i] = bf16(W[(c*FIN+i)*HID+o])   (transpose for B-frags)
// grid: 768 x 256
// =====================================================================
__global__ void k_wT(const float* __restrict__ W, unsigned short* __restrict__ wT) {
    int idx = blockIdx.x * 256 + threadIdx.x;        // < NL*NH*HID*FIN = 196608
    int c = idx >> 14;
    int r = idx & 16383;
    int o = r >> 8;
    int i = r & 255;
    wT[idx] = f2bf(W[((size_t)c * FIN + i) * HID + o]);
}

// =====================================================================
// e_score planes: esc[col][r] = sum_f e[r][f] * coef[colbase+col][f]
// r = n*GN + m flattened; one WMMA per 16 rows (K = 32 = EF exactly).
// grid: 2048 x 256 (8 waves/block, 16 tiles/wave, exact cover of PS/16 tiles)
// =====================================================================
#define ESC_TPW 16
__global__ void k_escore(const float* __restrict__ e, const float* __restrict__ coef,
                         float* __restrict__ esc, int ncols, int colbase) {
    const int gtid = blockIdx.x * blockDim.x + threadIdx.x;
    const int wave = gtid >> 5;
    const int lane = gtid & 31;
    const int l15  = lane & 15;
    const bool hiL = lane >= 16;
    // B fragment: lane(0-15) = column c holds K=0..15; lane(16-31) holds K=16..31
    v16bf bfrag;
    #pragma unroll
    for (int j = 0; j < 16; ++j) {
        float v = (l15 < ncols) ? coef[(colbase + l15) * EF + (hiL ? 16 : 0) + j] : 0.f;
        bfrag[j] = (__bf16)v;
    }
    const int k0 = hiL ? 16 : 0;                 // A frag K groups {0..15}/{16..31}
    for (int t = 0; t < ESC_TPW; ++t) {
        size_t rowbase = ((size_t)wave * ESC_TPW + t) * 16;
        if (rowbase >= PS) return;
        const float* ar = e + (rowbase + l15) * EF;
        float4 p0 = *(const float4*)(ar + k0);
        float4 p1 = *(const float4*)(ar + k0 + 4);
        float4 q0 = *(const float4*)(ar + k0 + 8);
        float4 q1 = *(const float4*)(ar + k0 + 12);
        v8f c = {};
        c = wmma_bf16(pack16(p0, p1, q0, q1), bfrag, c);
        if (l15 < ncols) {
            // C layout: VGPR v -> row M=v (lanes 0-15) / v+8 (lanes 16-31), col N=l15
            float* dst = esc + (size_t)l15 * PS + rowbase + (hiL ? 8 : 0);
            float4 lo = { c[0], c[1], c[2], c[3] };
            float4 hi = { c[4], c[5], c[6], c[7] };
            *(float4*)(dst)     = lo;
            *(float4*)(dst + 4) = hi;
        }
    }
}

// =====================================================================
// h = x @ W   (per head).  Writes hT[(hd*HID+o)*GN + n] in bf16.
// grid: (GN/16, NH), block 128 (4 waves; wave w -> output cols [16w,16w+16))
// =====================================================================
__global__ void k_gemm_xw(const float* __restrict__ x, const unsigned short* __restrict__ wTl,
                          unsigned short* __restrict__ hT) {
    const int hd   = blockIdx.y;
    const int tile = blockIdx.x;
    const int wave = threadIdx.x >> 5;
    const int lane = threadIdx.x & 31;
    const int l15  = lane & 15;
    const bool hiL = lane >= 16;
    const int oc0  = wave * 16;

    const int row = tile * 16 + l15;
    const float* xr = x + (size_t)row * FIN + (hiL ? 16 : 0);
    const unsigned short* wb =
        wTl + ((size_t)hd * HID + oc0 + l15) * FIN + (hiL ? 16 : 0);

    v8f c = {};
    #pragma unroll
    for (int ks = 0; ks < FIN / 32; ++ks) {
        const float* xk = xr + ks * 32;
        float4 p0 = *(const float4*)(xk);
        float4 p1 = *(const float4*)(xk + 4);
        float4 q0 = *(const float4*)(xk + 8);
        float4 q1 = *(const float4*)(xk + 12);
        v16u bu = *(const v16u*)(wb + ks * 32);
        c = wmma_bf16(pack16(p0, p1, q0, q1), __builtin_bit_cast(v16bf, bu), c);
    }
    const int o  = oc0 + l15;
    const int nb = tile * 16 + (hiL ? 8 : 0);
    v8bf outp;
    #pragma unroll
    for (int j = 0; j < 8; ++j) outp[j] = (__bf16)c[j];
    *(v8bf*)(hT + ((size_t)hd * HID + o) * GN + nb) = outp;
}

// =====================================================================
// s1[hd][n] = h[hd][n]·a1,  s2[hd][n] = h[hd][n]·a2
// grid: (GN/256, NH) x 256
// =====================================================================
__global__ void k_s(const unsigned short* __restrict__ hT, const float* __restrict__ a_l,
                    float* __restrict__ s1, float* __restrict__ s2) {
    const int hd = blockIdx.y;
    const int n  = blockIdx.x * 256 + threadIdx.x;
    const float* av = a_l + (size_t)hd * 192;
    float v1 = 0.f, v2 = 0.f;
    #pragma unroll 8
    for (int o = 0; o < HID; ++o) {
        float hv = bf2f(hT[((size_t)hd * HID + o) * GN + n]);
        v1 += hv * av[o];
        v2 += hv * av[64 + o];
    }
    s1[hd * GN + n] = v1;
    s2[hd * GN + n] = v2;
}

// =====================================================================
// Row softmax: att[hd][n][m] = softmax_m( mask ? leaky(s1+s2+esc) : -inf )
// One block per row n (all 4 heads share the adj row read). grid: GN x 256
// =====================================================================
__global__ void k_softmax(const float* __restrict__ esc, const float* __restrict__ s1,
                          const float* __restrict__ s2, const int* __restrict__ adj,
                          unsigned short* __restrict__ att) {
    const int n   = blockIdx.x;
    const int tid = threadIdx.x;
    __shared__ float red[256];

    float s1v[NH];
    #pragma unroll
    for (int hd = 0; hd < NH; ++hd) s1v[hd] = s1[hd * GN + n];

    float sc[NH][8];
    float lmax[NH];
    #pragma unroll
    for (int hd = 0; hd < NH; ++hd) lmax[hd] = -3.0e38f;

    #pragma unroll
    for (int j = 0; j < 8; ++j) {
        const int m = tid + j * 256;
        const int av = adj[(size_t)n * GN + m];
        #pragma unroll
        for (int hd = 0; hd < NH; ++hd) {
            float s = s1v[hd] + s2[hd * GN + m] + esc[(size_t)hd * PS + (size_t)n * GN + m];
            s = (s > 0.f) ? s : GALPHA * s;
            s = (av > 0) ? s : -1.0e30f;
            sc[hd][j] = s;
            lmax[hd] = fmaxf(lmax[hd], s);
        }
    }
    float mx[NH], inv[NH];
    for (int hd = 0; hd < NH; ++hd) {
        red[tid] = lmax[hd]; __syncthreads();
        for (int st = 128; st > 0; st >>= 1) {
            if (tid < st) red[tid] = fmaxf(red[tid], red[tid + st]);
            __syncthreads();
        }
        mx[hd] = red[0]; __syncthreads();
    }
    float lsum[NH] = {0.f, 0.f, 0.f, 0.f};
    #pragma unroll
    for (int j = 0; j < 8; ++j)
        #pragma unroll
        for (int hd = 0; hd < NH; ++hd) {
            sc[hd][j] = __expf(sc[hd][j] - mx[hd]);
            lsum[hd] += sc[hd][j];
        }
    for (int hd = 0; hd < NH; ++hd) {
        red[tid] = lsum[hd]; __syncthreads();
        for (int st = 128; st > 0; st >>= 1) {
            if (tid < st) red[tid] += red[tid + st];
            __syncthreads();
        }
        inv[hd] = 1.0f / red[0]; __syncthreads();
    }
    #pragma unroll
    for (int j = 0; j < 8; ++j)
        #pragma unroll
        for (int hd = 0; hd < NH; ++hd)
            att[((size_t)hd * GN + n) * GN + tid + j * 256] = f2bf(sc[hd][j] * inv[hd]);
}

// =====================================================================
// h' = att @ h  (per head).  Writes hpT[(hd*HID+o)*GN + n] in f32.
// grid: (GN/16, NH), block 128 (4 waves -> 4 x 16 output cols). K = 2048.
// =====================================================================
__global__ void k_gemm_att(const unsigned short* __restrict__ att,
                           const unsigned short* __restrict__ hT,
                           float* __restrict__ hpT) {
    const int hd   = blockIdx.y;
    const int tile = blockIdx.x;
    const int wave = threadIdx.x >> 5;
    const int lane = threadIdx.x & 31;
    const int l15  = lane & 15;
    const bool hiL = lane >= 16;
    const int oc0  = wave * 16;

    const int row = tile * 16 + l15;
    const unsigned short* arow = att + ((size_t)hd * GN + row) * GN;
    const unsigned short* hb   = hT + ((size_t)hd * HID + oc0 + l15) * GN + (hiL ? 16 : 0);
    const int k0 = hiL ? 16 : 0;

    v8f c = {};
    #pragma unroll 4
    for (int ks = 0; ks < GN / 32; ++ks) {
        const int kb = ks * 32;
        v8u a0 = *(const v8u*)(arow + kb + k0);        // K {0..7} / {16..23}
        v8u a1 = *(const v8u*)(arow + kb + k0 + 8);    // K {8..15} / {24..31}
        v16u au;
        #pragma unroll
        for (int j = 0; j < 8; ++j) { au[j] = a0[j]; au[8 + j] = a1[j]; }
        v16u bu = *(const v16u*)(hb + kb);
        c = wmma_bf16(__builtin_bit_cast(v16bf, au), __builtin_bit_cast(v16bf, bu), c);
    }
    float* dst = hpT + ((size_t)hd * HID + oc0 + l15) * GN + tile * 16 + (hiL ? 8 : 0);
    float4 lo = { c[0], c[1], c[2], c[3] };
    float4 hi = { c[4], c[5], c[6], c[7] };
    *(float4*)(dst)     = lo;
    *(float4*)(dst + 4) = hi;
}

// =====================================================================
// Epilogues: ELU + head-concat (mid layers) / head-mean (last layer)
// =====================================================================
__global__ void k_ep_mid(const float* __restrict__ hpT, float* __restrict__ xnext) {
    const int col = blockIdx.y;                         // 0..255 == hd*64+o
    const int n   = blockIdx.x * 256 + threadIdx.x;
    float v = hpT[(size_t)col * GN + n];
    v = (v > 0.f) ? v : (__expf(v) - 1.0f);
    xnext[(size_t)n * FIN + col] = v;
}
__global__ void k_ep_last(const float* __restrict__ hpT, float* __restrict__ out) {
    const int o = blockIdx.y;                           // 0..63
    const int n = blockIdx.x * 256 + threadIdx.x;
    float v = 0.f;
    #pragma unroll
    for (int hd = 0; hd < NH; ++hd) v += hpT[((size_t)hd * HID + o) * GN + n];
    v *= 0.25f;
    v = (v > 0.f) ? v : (__expf(v) - 1.0f);
    out[(size_t)n * HID + o] = v;
}

// =====================================================================
// Host orchestration
// =====================================================================
extern "C" void kernel_launch(void* const* d_in, const int* in_sizes, int n_in,
                              void* d_out, int out_size, void* d_ws, size_t ws_size,
                              hipStream_t stream) {
    (void)in_sizes; (void)n_in; (void)out_size;
    const float* x   = (const float*)d_in[0];
    const int*   adj = (const int*)d_in[1];
    const float* e   = (const float*)d_in[2];
    const float* We  = (const float*)d_in[3];
    const float* W   = (const float*)d_in[4];
    const float* a   = (const float*)d_in[5];
    float* out = (float*)d_out;
    char*  ws  = (char*)d_ws;

    // workspace layout (256B aligned)
    size_t off = 0;
    auto alloc = [&](size_t bytes) { size_t r = off; off = (off + bytes + 255) & ~(size_t)255; return r; };
    const size_t o_coef = alloc(16 * EF * sizeof(float));                      // padded to 16 cols
    const size_t o_wT   = alloc((size_t)NL * NH * HID * FIN * sizeof(unsigned short));
    const size_t o_hT   = alloc((size_t)NH * HID * GN * sizeof(unsigned short));
    const size_t o_s1   = alloc((size_t)NH * GN * sizeof(float));
    const size_t o_s2   = alloc((size_t)NH * GN * sizeof(float));
    const size_t o_hpT  = alloc((size_t)NH * HID * GN * sizeof(float));
    const size_t o_xA   = alloc((size_t)GN * FIN * sizeof(float));
    const size_t o_xB   = alloc((size_t)GN * FIN * sizeof(float));
    const size_t o_att  = alloc((size_t)NH * PS * sizeof(unsigned short));
    const size_t o_esc  = off;
    const bool all_mode = (ws_size >= o_esc + (size_t)NL * NH * PS * sizeof(float));

    float*          coef = (float*)(ws + o_coef);
    unsigned short* wT   = (unsigned short*)(ws + o_wT);
    unsigned short* hT   = (unsigned short*)(ws + o_hT);
    float*          s1   = (float*)(ws + o_s1);
    float*          s2   = (float*)(ws + o_s2);
    float*          hpT  = (float*)(ws + o_hpT);
    float*          xA   = (float*)(ws + o_xA);
    float*          xB   = (float*)(ws + o_xB);
    unsigned short* att  = (unsigned short*)(ws + o_att);
    float*          esc  = (float*)(ws + o_esc);

    hipMemsetAsync(coef, 0, 16 * EF * sizeof(float), stream);
    k_coef<<<dim3(NL * NH), dim3(256), 0, stream>>>(W, We, a, coef);
    k_wT<<<dim3((NL * NH * HID * FIN) / 256), dim3(256), 0, stream>>>(W, wT);

    if (all_mode)   // single pass over the 512MB e tensor for all 12 (l,h) planes
        k_escore<<<dim3(2048), dim3(256), 0, stream>>>(e, coef, esc, NL * NH, 0);

    for (int l = 0; l < NL; ++l) {
        const float* xin = (l == 0) ? x : ((l == 1) ? xA : xB);
        k_gemm_xw<<<dim3(GN / 16, NH), dim3(128), 0, stream>>>(
            xin, wT + (size_t)l * NH * HID * FIN, hT);
        k_s<<<dim3(GN / 256, NH), dim3(256), 0, stream>>>(hT, a + (size_t)l * NH * 192, s1, s2);
        if (!all_mode)
            k_escore<<<dim3(2048), dim3(256), 0, stream>>>(e, coef, esc, NH, l * NH);
        const float* esc_l = all_mode ? (esc + (size_t)l * NH * PS) : esc;
        k_softmax<<<dim3(GN), dim3(256), 0, stream>>>(esc_l, s1, s2, adj, att);
        k_gemm_att<<<dim3(GN / 16, NH), dim3(128), 0, stream>>>(att, hT, hpT);
        if (l < NL - 1)
            k_ep_mid<<<dim3(GN / 256, FIN), dim3(256), 0, stream>>>(hpT, (l == 0) ? xA : xB);
        else
            k_ep_last<<<dim3(GN / 256, HID), dim3(256), 0, stream>>>(hpT, out);
    }
}
